// DynamicCausalNetwork_23218593202495
// MI455X (gfx1250) — matrix operand order repaired and checked by reading, not compiled, not run on hardware
//
#include <hip/hip_runtime.h>
#include <string.h>

// Problem constants (from reference): B=16, T=256, R=64, H=128
constexpr int Bq  = 16;
constexpr int Tq  = 256;
constexpr int Rq  = 64;
constexpr int Hq  = 128;
constexpr int H3q = 384;

typedef __attribute__((ext_vector_type(16))) __bf16 v16bf;
typedef __attribute__((ext_vector_type(8)))  float  v8f;

__device__ __forceinline__ unsigned short f2bf(float f) {
  union { float f; unsigned u; } v; v.f = f;
  unsigned r = v.u + 0x7fffu + ((v.u >> 16) & 1u);  // round-to-nearest-even
  return (unsigned short)(r >> 16);
}

// A fragment (16x32 bf16): lane holds row=lane&15,
// K = kt*32 + (lane>>4)*16 + 0..15 -> 32 contiguous bytes.
__device__ __forceinline__ v16bf ldA(const unsigned short* p) {
  v16bf r;
  __builtin_memcpy(&r, p, 32);
  return r;
}
// B fragment (32x16 bf16, stored as W^T [j][k]): lane holds col=lane&15,
// K = kt*32 + (lane>>4)*16 + 0..15 -> 32 contiguous bytes.
__device__ __forceinline__ v16bf ldB(const unsigned short* p) {
  v16bf r;
  __builtin_memcpy(&r, p, 32);
  return r;
}

// Clamped Pade(7,6) tanh: 7 full-rate FMAs + one v_rcp. |err| < 1e-3 everywhere.
__device__ __forceinline__ float ptanh(float x) {
  x = fminf(fmaxf(x, -4.f), 4.f);
  float s = x * x;
  float num = x * fmaf(s, fmaf(s, (s + 378.f), 17325.f), 135135.f);
  float den = fmaf(s, fmaf(s, fmaf(s, 28.f, 3150.f), 62370.f), 135135.f);
  return num * __builtin_amdgcn_rcpf(den);
}
__device__ __forceinline__ float fsigmoid(float x) {
  return 1.f / (1.f + __expf(-x));
}

// Pre-transpose f32 weights [r][k][j] -> bf16 [r][j][k] so B-fragments are contiguous.
__global__ void k_transpose_bf16(const float* __restrict__ in,
                                 unsigned short* __restrict__ outw,
                                 int RR, int K, int J) {
  int idx = blockIdx.x * blockDim.x + threadIdx.x;
  int total = RR * K * J;
  if (idx >= total) return;
  int k  = idx % K;
  int j  = (idx / K) % J;
  int rr = idx / (K * J);
  outw[idx] = f2bf(in[((size_t)rr * K + k) * J + j]);
}

__global__ void k_zero(unsigned* p) {
  if (threadIdx.x == 0 && blockIdx.x == 0) *p = 0u;
}

#define WMMA_BF16(a, b, c) \
  __builtin_amdgcn_wmma_f32_16x16x32_bf16(false, (a), false, (b), (short)0, (c), false, false)

__global__ __launch_bounds__(512, 1) void dcn_persistent(
    const float* __restrict__ x,         // [B][T][R][H]
    const float* __restrict__ gbias,     // [R][2][3H]
    const float* __restrict__ causal_w,  // [R][R-1][H]
    const float* __restrict__ dense_b,   // [R][H]
    const unsigned short* __restrict__ Wk,  // bf16 [R][3H][H]
    const unsigned short* __restrict__ Wr,  // bf16 [R][3H][H]
    const unsigned short* __restrict__ Wd,  // bf16 [R][H][H]
    float* __restrict__ out,             // [B][T][R][H] (also carries recurrent state)
    unsigned* __restrict__ bar)
{
  // ---- resident LDS: 327,168 B of the WGP's 327,680 ----
  __shared__ __attribute__((aligned(16))) unsigned short sWk[H3q * Hq];   // 96 KB
  __shared__ __attribute__((aligned(16))) unsigned short sWr[H3q * Hq];   // 96 KB
  __shared__ __attribute__((aligned(16))) unsigned short sWd[Hq * Hq];    // 32 KB
  __shared__ float sCw[Rq * Hq];                                          // 32 KB
  __shared__ float sBin[H3q];
  __shared__ float sBrec[H3q];
  __shared__ float sDb[Hq];
  __shared__ float sPrev[Bq * Hq];                                        // 8 KB
  __shared__ __attribute__((aligned(16))) unsigned short sPrevBf[Bq * Hq];// 4 KB
  __shared__ __attribute__((aligned(16))) unsigned short sXtBf[Bq * Hq];  // 4 KB
  __shared__ float sZR[Bq * 2 * Hq];                                      // 16 KB
  __shared__ float sXH[Bq * Hq];                                          // 8 KB
  __shared__ float sRH[Bq * Hq];                                          // 8 KB
  __shared__ float sInfl[Bq * Hq];                                        // 8 KB
  __shared__ __attribute__((aligned(16))) unsigned short sComb[Bq * Hq];  // 4 KB

  const int r    = blockIdx.x;
  const int tid  = threadIdx.x;
  const int wave = tid >> 5;
  const int lane = tid & 31;

  // ---- one-time weight staging into LDS ----
  {
    const uint4* g = (const uint4*)(Wk + (size_t)r * H3q * Hq);
    uint4* s = (uint4*)sWk;
    for (int i = tid; i < H3q * Hq / 8; i += 512) s[i] = g[i];
    g = (const uint4*)(Wr + (size_t)r * H3q * Hq);
    s = (uint4*)sWr;
    for (int i = tid; i < H3q * Hq / 8; i += 512) s[i] = g[i];
    g = (const uint4*)(Wd + (size_t)r * Hq * Hq);
    s = (uint4*)sWd;
    for (int i = tid; i < Hq * Hq / 8; i += 512) s[i] = g[i];
  }
  // Scatter causal_w row r into dense [r2][h] with zero diagonal.
  for (int i = tid; i < Rq * Hq; i += 512) {
    int r2 = i >> 7, h = i & (Hq - 1);
    float v = 0.f;
    if (r2 < r)      v = causal_w[((size_t)r * (Rq - 1) + r2)     * Hq + h];
    else if (r2 > r) v = causal_w[((size_t)r * (Rq - 1) + r2 - 1) * Hq + h];
    sCw[i] = v;
  }
  for (int i = tid; i < H3q; i += 512) {
    sBin[i]  = gbias[(size_t)(2 * r)     * H3q + i];
    sBrec[i] = gbias[(size_t)(2 * r + 1) * H3q + i];
  }
  for (int i = tid; i < Hq; i += 512) sDb[i] = dense_b[(size_t)r * Hq + i];
  __syncthreads();

  const int lc   = lane & 15;
  const int hi   = lane >> 4;
  const int arow = lc * Hq;        // A fragment row base (row = batch index)
  const int kA   = hi << 4;        // A fragment half-offset per lane group
  const int kB   = hi << 4;        // B fragment half-offset per lane group

  for (int t = 0; t < Tq; ++t) {
    // ---- stage x_t and prev (= out at t-1) for this region ----
    for (int q = tid; q < Bq * Hq; q += 512) {
      int b = q >> 7, hc = q & (Hq - 1);
      float xv = x[(((size_t)b * Tq + t) * Rq + r) * Hq + hc];
      sXtBf[q] = f2bf(xv);
      float pv = (t > 0) ? out[(((size_t)b * Tq + (t - 1)) * Rq + r) * Hq + hc] : 0.f;
      sPrev[q]   = pv;
      sPrevBf[q] = f2bf(pv);
    }
    __syncthreads();

    // ---- merged epoch: gate GEMMs (tensor pipe) + influence reduction (VALU).
    //      Both depend only on staged prev/x, so they share one barrier epoch;
    //      per-wave order alternates so WMMA and tanh co-execute across waves. ----
    auto do_gemm = [&]() {
      const int u0   = wave << 1;
      const int kind = (u0 < 16) ? 0 : ((u0 < 24) ? 1 : 2);
      const int n0a  = (kind == 0) ? (u0 << 4)
                     : (kind == 1) ? (256 + ((u0 - 16) << 4))
                                   : (256 + ((u0 - 24) << 4));
      const int n0b  = n0a + 16;
      v8f c0 = {};
      v8f c1 = {};
      const unsigned short* Am = (kind == 2) ? sPrevBf : sXtBf;
      const unsigned short* Bm = (kind == 2) ? sWr : sWk;
      const unsigned short* b0p = Bm + (n0a + lc) * Hq + kB;
      const unsigned short* b1p = Bm + (n0b + lc) * Hq + kB;
#pragma unroll
      for (int kt = 0; kt < 4; ++kt) {
        v16bf a  = ldA(Am + arow + kt * 32 + kA);
        v16bf b0 = ldB(b0p + kt * 32);
        v16bf b1 = ldB(b1p + kt * 32);
        c0 = WMMA_BF16(a, b0, c0);
        c1 = WMMA_BF16(a, b1, c1);
      }
      if (kind == 0) {  // fuse recurrent contribution for z/r gate columns
        const unsigned short* b0q = sWr + (n0a + lc) * Hq + kB;
        const unsigned short* b1q = sWr + (n0b + lc) * Hq + kB;
#pragma unroll
        for (int kt = 0; kt < 4; ++kt) {
          v16bf a  = ldA(sPrevBf + arow + kt * 32 + kA);
          v16bf b0 = ldB(b0q + kt * 32);
          v16bf b1 = ldB(b1q + kt * 32);
          c0 = WMMA_BF16(a, b0, c0);
          c1 = WMMA_BF16(a, b1, c1);
        }
        const int ca = n0a + lc, cb = n0b + lc;
        const float bsa = sBin[ca] + sBrec[ca];
        const float bsb = sBin[cb] + sBrec[cb];
#pragma unroll
        for (int i = 0; i < 8; ++i) {
          int m = i + (hi << 3);
          sZR[m * 256 + ca] = fsigmoid(c0[i] + bsa);
          sZR[m * 256 + cb] = fsigmoid(c1[i] + bsb);
        }
      } else if (kind == 1) {
        const int ca = n0a + lc, cb = n0b + lc;
        const float bsa = sBin[ca], bsb = sBin[cb];
#pragma unroll
        for (int i = 0; i < 8; ++i) {
          int m = i + (hi << 3);
          sXH[m * Hq + (ca - 256)] = c0[i] + bsa;
          sXH[m * Hq + (cb - 256)] = c1[i] + bsb;
        }
      } else {
        const int ca = n0a + lc, cb = n0b + lc;
        const float bsa = sBrec[ca], bsb = sBrec[cb];
#pragma unroll
        for (int i = 0; i < 8; ++i) {
          int m = i + (hi << 3);
          sRH[m * Hq + (ca - 256)] = c0[i] + bsa;
          sRH[m * Hq + (cb - 256)] = c1[i] + bsb;
        }
      }
    };

    auto do_infl = [&]() {
      // Each thread owns one h-column (hc) and four batch rows (b0+4p).
      const int hc = tid & (Hq - 1);
      const int b0 = tid >> 7;  // 0..3
      float a0 = 0.f, a1 = 0.f, a2 = 0.f, a3 = 0.f;
      if (t > 0) {
        const size_t bs = (size_t)Tq * Rq * Hq;
        const float* p0 = out + ((size_t)(b0 + 0)  * Tq + (t - 1)) * (Rq * Hq) + hc;
        const float* p1 = p0 + 4  * bs;
        const float* p2 = p0 + 8  * bs;
        const float* p3 = p0 + 12 * bs;
#pragma unroll 8
        for (int r2 = 0; r2 < Rq; ++r2) {
          float w = sCw[r2 * Hq + hc];              // shared by all 4 batch rows
          size_t off = (size_t)r2 * Hq;
          a0 += ptanh(w * p0[off]);
          a1 += ptanh(w * p1[off]);
          a2 += ptanh(w * p2[off]);
          a3 += ptanh(w * p3[off]);                 // diagonal cw==0 -> tanh(0)=0
        }
      }
      sInfl[(b0 + 0)  * Hq + hc] = a0;
      sInfl[(b0 + 4)  * Hq + hc] = a1;
      sInfl[(b0 + 8)  * Hq + hc] = a2;
      sInfl[(b0 + 12) * Hq + hc] = a3;
    };

    if (wave & 1) { do_gemm(); do_infl(); }
    else          { do_infl(); do_gemm(); }
    __syncthreads();

    // ---- gate combine (light): hh, h, combined = tanh(h + infl) ----
    for (int q = 0; q < 4; ++q) {
      int idx = tid + q * 512;            // 2048 (b,h) pairs
      int b = idx >> 7, hc = idx & (Hq - 1);
      float z  = sZR[b * 256 + hc];
      float rg = sZR[b * 256 + 128 + hc];
      float xh = sXH[b * Hq + hc];
      float rh = sRH[b * Hq + hc];
      float hh = ptanh(xh + rg * rh);
      float h  = z * sPrev[b * Hq + hc] + (1.f - z) * hh;
      sComb[b * Hq + hc] = f2bf(ptanh(h + sInfl[b * Hq + hc]));
    }
    __syncthreads();

    // ---- dense output GEMM (8 tiles, waves 0-7) ----
    if (wave < 8) {
      const int n0 = wave << 4;
      v8f c = {};
      const unsigned short* bp = sWd + (n0 + lc) * Hq + kB;
#pragma unroll
      for (int kt = 0; kt < 4; ++kt) {
        v16bf a = ldA(sComb + arow + kt * 32 + kA);
        v16bf b = ldB(bp + kt * 32);
        c = WMMA_BF16(a, b, c);
      }
      const int col = n0 + lc;
      const float db = sDb[col];
#pragma unroll
      for (int i = 0; i < 8; ++i) {
        int m = i + (hi << 3);
        out[(((size_t)m * Tq + t) * Rq + r) * Hq + col] = c[i] + db;
      }
    }

    // ---- device-wide step barrier: all 64 region blocks sync through L2 ----
    __threadfence();
    __syncthreads();
    if (tid == 0) {
      __hip_atomic_fetch_add(bar, 1u, __ATOMIC_ACQ_REL, __HIP_MEMORY_SCOPE_AGENT);
      unsigned target = (unsigned)(t + 1) * (unsigned)Rq;
      while (__hip_atomic_load(bar, __ATOMIC_ACQUIRE, __HIP_MEMORY_SCOPE_AGENT) < target)
        __builtin_amdgcn_s_sleep(2);
    }
    __syncthreads();
    __threadfence();
  }
}

extern "C" void kernel_launch(void* const* d_in, const int* in_sizes, int n_in,
                              void* d_out, int out_size, void* d_ws, size_t ws_size,
                              hipStream_t stream) {
  const float* x   = (const float*)d_in[0];  // inputs      [B][T][R][H]
  const float* gk  = (const float*)d_in[1];  // gru_kernel  [R][H][3H]
  const float* grk = (const float*)d_in[2];  // gru_rkernel [R][H][3H]
  const float* gb  = (const float*)d_in[3];  // gru_bias    [R][2][3H]
  const float* cw  = (const float*)d_in[4];  // causal_w    [R][R-1][H]
  const float* dw  = (const float*)d_in[5];  // dense_w     [R][H][H]
  const float* db  = (const float*)d_in[6];  // dense_b     [R][H]
  float* out = (float*)d_out;

  // Workspace layout: bf16 transposed weights + barrier counter (~14.7 MB).
  unsigned short* Wk = (unsigned short*)d_ws;
  unsigned short* Wr = Wk + (size_t)Rq * H3q * Hq;
  unsigned short* Wd = Wr + (size_t)Rq * H3q * Hq;
  unsigned* bar = (unsigned*)(Wd + (size_t)Rq * Hq * Hq);

  int n1 = Rq * Hq * H3q;
  k_transpose_bf16<<<(n1 + 255) / 256, 256, 0, stream>>>(gk,  Wk, Rq, Hq, H3q);
  k_transpose_bf16<<<(n1 + 255) / 256, 256, 0, stream>>>(grk, Wr, Rq, Hq, H3q);
  int n2 = Rq * Hq * Hq;
  k_transpose_bf16<<<(n2 + 255) / 256, 256, 0, stream>>>(dw,  Wd, Rq, Hq, Hq);
  k_zero<<<1, 1, 0, stream>>>(bar);

  dcn_persistent<<<Rq, 512, 0, stream>>>(x, gb, cw, db, Wk, Wr, Wd, out, bar);
}
